// Ner_Fusion_Feature_60206851555512
// MI455X (gfx1250) — matrix operand order repaired and checked by reading, not compiled, not run on hardware
//
#include <hip/hip_runtime.h>
#include <math.h>

// ---------------------------------------------------------------------------
// NER fusion feature pipeline for MI455X (gfx1250, wave32, WMMA).
//
// Exact algebraic simplifications vs the reference (see analysis):
//  * softmax(q + k_j + c) over j == softmax(k_j): both attention maps are
//    query-independent -> att_img / att_text are rank-1 per batch (ctx1/ctx2).
//    t1_w/t1_b, i2_w/i2_b, a1_w[:H], a2_w[:H], a1_b, a2_b are dead.
//  * fg_w is (2H,1): (X@W)@v == X@(W@v) -> ft_w/fm_w GEMMs fold to vectors.
//
// Surviving heavy work: k_sc = tanh(text@t2_w)@a2_w[H:]  (19.3 GFLOP, WMMA)
//                       i_sc = tanh(img @i1_w)@a1_w[H:]  ( 1.9 GFLOP, WMMA)
// Pipeline is HBM-bound: ~100 MB reads of text + 50 MB output ~= 6-7 us.
// ---------------------------------------------------------------------------

static constexpr int kH  = 768;
static constexpr int kB  = 32;
static constexpr int kS  = 512;
static constexpr int kR  = 49;
static constexpr int kBS = kB * kS;   // 16384 text rows
static constexpr int kBR = kB * kR;   // 1568 img rows

typedef __bf16 bf16;
typedef __attribute__((ext_vector_type(16))) __bf16 v16bf;
typedef __attribute__((ext_vector_type(8)))  __bf16 v8bf;
typedef __attribute__((ext_vector_type(8)))  float  v8f;

union Frag16 { v16bf v; v8bf h[2]; };

// ---- prep: W (KxN row-major fp32) -> wT (NxK row-major bf16) ---------------
__global__ void k_prep_transpose(const float* __restrict__ w, bf16* __restrict__ wT) {
  int i = blockIdx.x * blockDim.x + threadIdx.x;
  if (i >= kH * kH) return;
  int n = i / kH, k = i % kH;
  wT[i] = (bf16)w[k * kH + n];
}

// vft[h] = sum_j ft_w[h][j]*fg_w[j];  vfm[h] = sum_j fm_w[h][j]*fg_w[H+j]
__global__ void k_prep_vec(const float* __restrict__ ftw, const float* __restrict__ fmw,
                           const float* __restrict__ fgw, float* __restrict__ vft,
                           float* __restrict__ vfm) {
  int h = blockIdx.x * blockDim.x + threadIdx.x;
  if (h >= kH) return;
  float s0 = 0.f, s1 = 0.f;
  for (int j = 0; j < kH; ++j) {
    s0 += ftw[h * kH + j] * fgw[j];
    s1 += fmw[h * kH + j] * fgw[kH + j];
  }
  vft[h] = s0;
  vfm[h] = s1;
}

// ---------------------------------------------------------------------------
// WMMA GEMM + fused tanh-weighted row reduction:
//   outRed[m] = sum_n tanh( sum_k A[m,k] * W[k,n] ) * av[n]
// optionally fused GEMV: outTf[m] = sum_k A[m,k] * vft[k]  (reads A once).
// A: M x 768 fp32 (M = 16*gridDim.x), wT: 768x768 bf16 N-major.
// Block = 8 waves; BM=16; wave w owns N in [96w, 96w+96) = 6 WMMA tiles.
// ---------------------------------------------------------------------------
static constexpr int kLdsStride = 776;  // 768 + 8 bf16 pad (conflict-free b128 rows)

__global__ __launch_bounds__(256) void k_gemm_tanh_reduce(
    const float* __restrict__ A, const bf16* __restrict__ wT,
    const float* __restrict__ av, float* __restrict__ outRed,
    const float* __restrict__ vft, float* __restrict__ outTf) {
  __shared__ bf16  As[16 * kLdsStride];
  __shared__ float redpart[8][16];
  __shared__ float tfpart[256];

  const int tid   = threadIdx.x;
  const int lane  = tid & 31;
  const int wv    = tid >> 5;          // wave 0..7
  const int mBase = blockIdx.x * 16;

  // ---- cooperative load: 16 rows x 768 fp32 -> bf16 LDS (+ fused A.vft) ----
  {
    const int   row = tid >> 4;        // 16 threads per row
    const int   cl  = tid & 15;
    const float4* Ar = (const float4*)(A + (size_t)(mBase + row) * kH);
    float s = 0.f;
#pragma unroll
    for (int j = 0; j < 12; ++j) {
      const int c4 = cl + j * 16;      // 192 float4 per row
      float4 x = Ar[c4];
      bf16* dst = &As[row * kLdsStride + c4 * 4];
      dst[0] = (bf16)x.x; dst[1] = (bf16)x.y; dst[2] = (bf16)x.z; dst[3] = (bf16)x.w;
      if (vft) {
        const float* vp = vft + c4 * 4;
        s += x.x * vp[0] + x.y * vp[1] + x.z * vp[2] + x.w * vp[3];
      }
    }
    tfpart[tid] = s;
  }
  __syncthreads();

  // ---- WMMA main loop ------------------------------------------------------
  const int nb  = wv * 96;
  const int row = lane & 15;           // A/B: M index / N index within tile
  const int hi  = lane >> 4;           // lane half selects K sub-range
  float avj[6];
#pragma unroll
  for (int j = 0; j < 6; ++j) avj[j] = av[nb + j * 16 + row];

  v8f acc[6] = {};
  for (int kt = 0; kt < 24; ++kt) {
    const int k0 = kt * 32;
    // A fragment (16x32 bf16): lanes 0-15: K {k0..k0+7, k0+16..k0+23};
    //                          lanes 16-31: K {k0+8..k0+15, k0+24..k0+31}.
    Frag16 a;
    const bf16* ap = &As[row * kLdsStride + k0 + hi * 8];
    a.h[0] = *(const v8bf*)ap;
    a.h[1] = *(const v8bf*)(ap + 16);
#pragma unroll
    for (int j = 0; j < 6; ++j) {
      // B fragment (32x16 bf16): lanes 0-15: K {k0..k0+7, k0+16..k0+23};
      // lanes 16-31: K {k0+8..k0+15, k0+24..k0+31} of column n -> wT is N-major.
      Frag16 b;
      const bf16* bp = wT + (size_t)(nb + j * 16 + row) * kH + k0 + hi * 8;
      b.h[0] = *(const v8bf*)bp;
      b.h[1] = *(const v8bf*)(bp + 16);
      acc[j] = __builtin_amdgcn_wmma_f32_16x16x32_bf16(
          false, a.v, false, b.v, (short)0, acc[j], false, false);
    }
  }

  // ---- epilogue: rowsum[v] = sum_j tanh(acc[j][v]) * av[n_j] ---------------
  float rowsum[8];
#pragma unroll
  for (int v = 0; v < 8; ++v) rowsum[v] = 0.f;
#pragma unroll
  for (int j = 0; j < 6; ++j) {
    const float w = avj[j];
#pragma unroll
    for (int v = 0; v < 8; ++v) rowsum[v] += tanhf(acc[j][v]) * w;
  }
  // butterfly across the 16 lanes sharing the same row group (deterministic)
#pragma unroll
  for (int m = 1; m < 16; m <<= 1)
#pragma unroll
    for (int v = 0; v < 8; ++v) rowsum[v] += __shfl_xor(rowsum[v], m, 32);
  if ((lane & 15) == 0) {
    const int base = hi * 8;           // C layout: m = v + 8*(lane>=16)
#pragma unroll
    for (int v = 0; v < 8; ++v) redpart[wv][base + v] = rowsum[v];
  }
  __syncthreads();

  if (tid < 16) {
    float r = 0.f;
#pragma unroll
    for (int w = 0; w < 8; ++w) r += redpart[w][tid];
    outRed[mBase + tid] = r;
    if (outTf) {
      float t = 0.f;
#pragma unroll
      for (int i = 0; i < 16; ++i) t += tfpart[tid * 16 + i];
      outTf[mBase + tid] = t;
    }
  }
}

// ---- softmax over N scores per batch + weighted row-sum ctx[b,:] -----------
__global__ __launch_bounds__(256) void k_softmax_ctx(
    const float* __restrict__ sc, int N, const float* __restrict__ X,
    float* __restrict__ ctx) {
  const int b = blockIdx.x, tid = threadIdx.x;
  sc  += (size_t)b * N;
  X   += (size_t)b * N * kH;
  ctx += (size_t)b * kH;
  __shared__ float p[512];
  __shared__ float r[256];

  float m = -INFINITY;
  for (int j = tid; j < N; j += 256) m = fmaxf(m, sc[j]);
  r[tid] = m; __syncthreads();
  for (int s = 128; s > 0; s >>= 1) {
    if (tid < s) r[tid] = fmaxf(r[tid], r[tid + s]);
    __syncthreads();
  }
  const float mx = r[0];
  __syncthreads();

  float s = 0.f;
  for (int j = tid; j < N; j += 256) { float e = __expf(sc[j] - mx); p[j] = e; s += e; }
  r[tid] = s; __syncthreads();
  for (int st = 128; st > 0; st >>= 1) {
    if (tid < st) r[tid] += r[tid + st];
    __syncthreads();
  }
  const float inv = 1.f / r[0];

  float a0 = 0.f, a1 = 0.f, a2 = 0.f;
  for (int j = 0; j < N; ++j) {
    const float w = p[j] * inv;
    const float* xr = X + (size_t)j * kH;
    a0 += w * xr[tid];
    a1 += w * xr[tid + 256];
    a2 += w * xr[tid + 512];
  }
  ctx[tid] = a0; ctx[tid + 256] = a1; ctx[tid + 512] = a2;
}

// ---- tiny M=32 GEMM: out[b,n] = tanh( sum_k x[b,k]*W[k,n] + bias[n] ) ------
__global__ __launch_bounds__(256) void k_small_gemm_tanh(
    const float* __restrict__ x, const float* __restrict__ W,
    const float* __restrict__ bias, float* __restrict__ out) {
  const int n = blockIdx.x * 256 + threadIdx.x;
  const int b = blockIdx.y;
  const float* xr = x + (size_t)b * kH;
  float acc = bias[n];
  for (int k = 0; k < kH; ++k) acc += xr[k] * W[(size_t)k * kH + n];
  out[(size_t)b * kH + n] = tanhf(acc);
}

// ---- GMF gate + multimodal blend (per batch) -------------------------------
__global__ __launch_bounds__(256) void k_gate_mm(
    const float* __restrict__ ni, const float* __restrict__ nt,
    const float* __restrict__ ggw, const float* __restrict__ ggb,
    float* __restrict__ mm) {
  const int b = blockIdx.x, tid = threadIdx.x;
  const float* nib = ni + (size_t)b * kH;
  const float* ntb = nt + (size_t)b * kH;
  __shared__ float r[256];
  float s = 0.f;
  for (int h = tid; h < kH; h += 256) s += nib[h] * ggw[h] + ntb[h] * ggw[kH + h];
  r[tid] = s; __syncthreads();
  for (int st = 128; st > 0; st >>= 1) {
    if (tid < st) r[tid] += r[tid + st];
    __syncthreads();
  }
  const float g = 1.f / (1.f + __expf(-(r[0] + ggb[0])));
  for (int h = tid; h < kH; h += 256)
    mm[(size_t)b * kH + h] = g * nib[h] + (1.f - g) * ntb[h];
}

// ---- per-batch filtration-gate constant: cb[b] = mm.vfm + fm_b.fg_w[H:]+fg_b
__global__ __launch_bounds__(256) void k_cb(
    const float* __restrict__ mm, const float* __restrict__ vfm,
    const float* __restrict__ fmb, const float* __restrict__ fgw,
    const float* __restrict__ fgb, float* __restrict__ cb) {
  const int b = blockIdx.x, tid = threadIdx.x;
  __shared__ float r[256];
  float s = 0.f;
  for (int h = tid; h < kH; h += 256)
    s += mm[(size_t)b * kH + h] * vfm[h] + fmb[h] * fgw[kH + h];
  r[tid] = s; __syncthreads();
  for (int st = 128; st > 0; st >>= 1) {
    if (tid < st) r[tid] += r[tid + st];
    __syncthreads();
  }
  if (tid == 0) cb[b] = r[0] + fgb[0];
}

// ---- output: out[b,s,:] = sigmoid(tf[b,s] + cb[b]) * u[b,:] ----------------
__global__ __launch_bounds__(256) void k_out(
    const float* __restrict__ tf, const float* __restrict__ cb,
    const float* __restrict__ u, float* __restrict__ out) {
  const int i   = blockIdx.x * 256 + threadIdx.x;   // over BS * 192 float4
  const int row = i / 192;
  const int c4  = i % 192;
  const int b   = row / kS;
  const float fg = 1.f / (1.f + __expf(-(tf[row] + cb[b])));
  const float4 uv = ((const float4*)u)[(size_t)b * 192 + c4];
  float4 o; o.x = fg * uv.x; o.y = fg * uv.y; o.z = fg * uv.z; o.w = fg * uv.w;
  ((float4*)out)[i] = o;
}

// ---------------------------------------------------------------------------
extern "C" void kernel_launch(void* const* d_in, const int* in_sizes, int n_in,
                              void* d_out, int out_size, void* d_ws, size_t ws_size,
                              hipStream_t stream) {
  (void)in_sizes; (void)n_in; (void)out_size; (void)ws_size;
  const float* text = (const float*)d_in[0];
  const float* img  = (const float*)d_in[1];
  // d_in[2]=t1_w, [3]=t1_b, [6]=a1_b, [8]=i2_w, [9]=i2_b, [11]=a2_b: dead code.
  const float* i1_w = (const float*)d_in[4];
  const float* a1_w = (const float*)d_in[5];
  const float* t2_w = (const float*)d_in[7];
  const float* a2_w = (const float*)d_in[10];
  const float* gt_w = (const float*)d_in[12];
  const float* gt_b = (const float*)d_in[13];
  const float* gi_w = (const float*)d_in[14];
  const float* gi_b = (const float*)d_in[15];
  const float* gg_w = (const float*)d_in[16];
  const float* gg_b = (const float*)d_in[17];
  const float* ft_w = (const float*)d_in[18];
  const float* fm_w = (const float*)d_in[19];
  const float* fm_b = (const float*)d_in[20];
  const float* fg_w = (const float*)d_in[21];
  const float* fg_b = (const float*)d_in[22];
  const float* fr_w = (const float*)d_in[23];
  const float* fr_b = (const float*)d_in[24];

  char* ws = (char*)d_ws;
  size_t off = 0;
  auto carve = [&](size_t bytes) {
    void* p = ws + off;
    off += (bytes + 255) & ~(size_t)255;
    return p;
  };
  bf16*  t2T  = (bf16*)carve((size_t)kH * kH * sizeof(bf16));
  bf16*  i1T  = (bf16*)carve((size_t)kH * kH * sizeof(bf16));
  float* k_sc = (float*)carve((size_t)kBS * sizeof(float));
  float* i_sc = (float*)carve((size_t)kBR * sizeof(float));
  float* tf   = (float*)carve((size_t)kBS * sizeof(float));
  float* ctx1 = (float*)carve((size_t)kB * kH * sizeof(float));
  float* ctx2 = (float*)carve((size_t)kB * kH * sizeof(float));
  float* ni   = (float*)carve((size_t)kB * kH * sizeof(float));
  float* nt   = (float*)carve((size_t)kB * kH * sizeof(float));
  float* mm   = (float*)carve((size_t)kB * kH * sizeof(float));
  float* u    = (float*)carve((size_t)kB * kH * sizeof(float));
  float* vft  = (float*)carve((size_t)kH * sizeof(float));
  float* vfm  = (float*)carve((size_t)kH * sizeof(float));
  float* cb   = (float*)carve((size_t)kB * sizeof(float));

  // prep: bf16-transposed weights + folded filtration vectors
  k_prep_transpose<<<(kH * kH + 255) / 256, 256, 0, stream>>>(t2_w, t2T);
  k_prep_transpose<<<(kH * kH + 255) / 256, 256, 0, stream>>>(i1_w, i1T);
  k_prep_vec<<<3, 256, 0, stream>>>(ft_w, fm_w, fg_w, vft, vfm);

  // the two surviving GEMMs (WMMA bf16, fused tanh-reduce epilogue)
  k_gemm_tanh_reduce<<<kBR / 16, 256, 0, stream>>>(img, i1T, a1_w + kH, i_sc,
                                                   nullptr, nullptr);
  k_gemm_tanh_reduce<<<kBS / 16, 256, 0, stream>>>(text, t2T, a2_w + kH, k_sc,
                                                   vft, tf);

  // query-independent attention contexts (rank-1 per batch)
  k_softmax_ctx<<<kB, 256, 0, stream>>>(i_sc, kR, img, ctx1);
  k_softmax_ctx<<<kB, 256, 0, stream>>>(k_sc, kS, text, ctx2);

  // GMF fusion + filtration on the 32 per-batch vectors
  dim3 g3(3, kB);
  k_small_gemm_tanh<<<g3, 256, 0, stream>>>(ctx1, gi_w, gi_b, ni);
  k_small_gemm_tanh<<<g3, 256, 0, stream>>>(ctx2, gt_w, gt_b, nt);
  k_gate_mm<<<kB, 256, 0, stream>>>(ni, nt, gg_w, gg_b, mm);
  k_small_gemm_tanh<<<g3, 256, 0, stream>>>(mm, fr_w, fr_b, u);
  k_cb<<<kB, 256, 0, stream>>>(mm, vfm, fm_b, fg_w, fg_b, cb);

  // broadcast-scaled output (50 MB stream write)
  k_out<<<kBS * 192 / 256, 256, 0, stream>>>(tf, cb, u, (float*)d_out);
}
